// SConvNetBlock_36696200577232
// MI455X (gfx1250) — matrix operand order repaired and checked by reading, not compiled, not run on hardware
//
#include <hip/hip_runtime.h>
#include <hip/hip_bf16.h>

// ---------------------------------------------------------------------------
// SConvNet block for MI455X (gfx1250, wave32, WMMA).
//   LN1 -> SiLU -> chunked complex scan (SConv) -> *sc_lin + residual
//   LN2 -> bf16 -> WMMA GEMM1(+b1, SiLU) -> WMMA GEMM2(+b2, +residual)
// GEMMs: v_wmma_f32_16x16x32_bf16, double-buffered LDS,
//        A-tile via GLOBAL_LOAD_ASYNC_TO_LDS (ASYNCcnt),
//        B-tile via Tensor Data Mover (TENSORcnt) when available.
// ---------------------------------------------------------------------------

#define DIMC 1024
#define FFC  4096
#define BATC 4
#define SEQC 2048
#define ROWSC (BATC * SEQC)     // 8192
#define NCH   16                // scan chunks per sequence
#define CLEN  128               // chunk length (NCH*CLEN == SEQC)
#define LDSK  40                // padded K stride in LDS (32 data + 8 pad bf16)

typedef __attribute__((ext_vector_type(16))) __bf16 bf16x16;
typedef __attribute__((ext_vector_type(8)))  __bf16 bf16x8;
typedef __attribute__((ext_vector_type(8)))  float  f32x8;
typedef unsigned int u32x4 __attribute__((ext_vector_type(4)));
typedef int          i32x4 __attribute__((ext_vector_type(4)));
typedef int          i32x8 __attribute__((ext_vector_type(8)));
// exact pointee type expected by __builtin_amdgcn_global_load_async_to_lds_b128
typedef int v4i_vs __attribute__((vector_size(4 * sizeof(int))));

// Feature probes -------------------------------------------------------------
#if defined(__has_builtin)
#  if __has_builtin(__builtin_amdgcn_global_load_async_to_lds_b128)
#    define HAVE_ASYNC_LDS 1
#  endif
#  if __has_builtin(__builtin_amdgcn_tensor_load_to_lds)
#    define HAVE_TDM 1
#  endif
#endif
#ifndef HAVE_ASYNC_LDS
#  define HAVE_ASYNC_LDS 0
#endif
#ifndef HAVE_TDM
#  define HAVE_TDM 0
#endif

// ---------------------------------------------------------------------------
// fp32 -> bf16 elementwise (weights)
// ---------------------------------------------------------------------------
__global__ __launch_bounds__(256)
void cvt_bf16_kernel(const float* __restrict__ src, __bf16* __restrict__ dst, int n) {
    int i = blockIdx.x * 256 + threadIdx.x;
    int stride = gridDim.x * 256;
    for (; i < n; i += stride) dst[i] = (__bf16)src[i];
}

// ---------------------------------------------------------------------------
// LayerNorm helpers (one block = one row of 1024)
// ---------------------------------------------------------------------------
__device__ __forceinline__ void row_stats(const float* __restrict__ xr,
                                          float& mean, float& inv) {
    float s = 0.f, s2 = 0.f;
    for (int i = threadIdx.x; i < DIMC; i += 256) {
        float v = xr[i]; s += v; s2 += v * v;
    }
#pragma unroll
    for (int off = 16; off > 0; off >>= 1) {
        s  += __shfl_down(s,  off, 32);
        s2 += __shfl_down(s2, off, 32);
    }
    __shared__ float rs[8], rq[8];
    const int wid = threadIdx.x >> 5;
    if ((threadIdx.x & 31) == 0) { rs[wid] = s; rq[wid] = s2; }
    __syncthreads();
    if (threadIdx.x == 0) {
        float t = 0.f, t2 = 0.f;
#pragma unroll
        for (int i = 0; i < 8; ++i) { t += rs[i]; t2 += rq[i]; }
        rs[0] = t; rq[0] = t2;
    }
    __syncthreads();
    mean = rs[0] * (1.0f / DIMC);
    float var = rq[0] * (1.0f / DIMC) - mean * mean;
    inv = rsqrtf(var + 1e-5f);
    __syncthreads();
}

__global__ __launch_bounds__(256)
void ln1_silu_kernel(const float* __restrict__ x, const float* __restrict__ w,
                     const float* __restrict__ b, float* __restrict__ out) {
    const int row = blockIdx.x;
    const float* xr = x + (size_t)row * DIMC;
    float mean, inv;
    row_stats(xr, mean, inv);
    for (int i = threadIdx.x; i < DIMC; i += 256) {
        float v = (xr[i] - mean) * inv * w[i] + b[i];
        v = v / (1.0f + __expf(-v));            // SiLU
        out[(size_t)row * DIMC + i] = v;
    }
}

__global__ __launch_bounds__(256)
void ln2_bf16_kernel(const float* __restrict__ x, const float* __restrict__ w,
                     const float* __restrict__ b, __bf16* __restrict__ out) {
    const int row = blockIdx.x;
    const float* xr = x + (size_t)row * DIMC;
    float mean, inv;
    row_stats(xr, mean, inv);
    for (int i = threadIdx.x; i < DIMC; i += 256) {
        float v = (xr[i] - mean) * inv * w[i] + b[i];
        out[(size_t)row * DIMC + i] = (__bf16)v;
    }
}

// ---------------------------------------------------------------------------
// SConv = first-order complex IIR:  s_t = p*s_{t-1} + h_t,  s_{-1} = last,
// out_t = Re(s_t) * sc_lin + x_t.   3-pass chunked scan (chunk = 128 steps).
// ---------------------------------------------------------------------------
__device__ __forceinline__ void make_p(const float* __restrict__ pre,
                                       const float* __restrict__ pim,
                                       int d, float& pr, float& pi) {
    float ar = pre[d], ai = pim[d];
    float mag = sqrtf(ar * ar + ai * ai) + 1e-20f;
    float t = tanhf(mag) / mag;
    pr = t * ar; pi = t * ai;
}

__global__ __launch_bounds__(256)
void scan_pass1_kernel(const float* __restrict__ h,
                       const float* __restrict__ pre, const float* __restrict__ pim,
                       float* __restrict__ chunkS) {
    const int tid = blockIdx.x * 256 + threadIdx.x;   // B*NCH*DIM threads
    const int d = tid & (DIMC - 1);
    const int c = (tid >> 10) & (NCH - 1);
    const int b = tid >> 14;
    float pr, pi; make_p(pre, pim, d, pr, pi);
    float sr = 0.f, si = 0.f;
    const float* hp = h + ((size_t)(b * SEQC + c * CLEN)) * DIMC + d;
    for (int i = 0; i < CLEN; ++i) {
        float hv = hp[(size_t)i * DIMC];
        float nr = pr * sr - pi * si + hv;
        float ni = pr * si + pi * sr;
        sr = nr; si = ni;
    }
    chunkS[2 * (size_t)tid]     = sr;
    chunkS[2 * (size_t)tid + 1] = si;
}

__global__ __launch_bounds__(256)
void scan_pass2_kernel(const float* __restrict__ pre, const float* __restrict__ pim,
                       const float* __restrict__ last_re, const float* __restrict__ last_im,
                       const float* __restrict__ chunkS, float* __restrict__ chunkI) {
    const int tid = blockIdx.x * 256 + threadIdx.x;   // B*DIM threads
    const int d = tid & (DIMC - 1);
    const int b = tid >> 10;
    float pr, pi; make_p(pre, pim, d, pr, pi);
    float qr = pr, qi = pi;                           // p^(2^k) -> p^128
#pragma unroll
    for (int k = 0; k < 7; ++k) {
        float nr = qr * qr - qi * qi;
        float ni = 2.0f * qr * qi;
        qr = nr; qi = ni;
    }
    float cr = last_re[d], ci = last_im[d];
    for (int c = 0; c < NCH; ++c) {
        const size_t idx = ((size_t)b * NCH + c) * DIMC + d;
        chunkI[2 * idx]     = cr;
        chunkI[2 * idx + 1] = ci;
        float sr = chunkS[2 * idx], si = chunkS[2 * idx + 1];
        float nr = qr * cr - qi * ci + sr;
        float ni = qr * ci + qi * cr + si;
        cr = nr; ci = ni;
    }
}

__global__ __launch_bounds__(256)
void scan_pass3_kernel(const float* __restrict__ h, const float* __restrict__ x,
                       const float* __restrict__ pre, const float* __restrict__ pim,
                       const float* __restrict__ sc_lin,
                       const float* __restrict__ chunkI, float* __restrict__ x1) {
    const int tid = blockIdx.x * 256 + threadIdx.x;
    const int d = tid & (DIMC - 1);
    const int c = (tid >> 10) & (NCH - 1);
    const int b = tid >> 14;
    float pr, pi; make_p(pre, pim, d, pr, pi);
    float sr = chunkI[2 * (size_t)tid], si = chunkI[2 * (size_t)tid + 1];
    const float sc = sc_lin[d];
    const size_t base = ((size_t)(b * SEQC + c * CLEN)) * DIMC + d;
    const float* hp = h + base;
    const float* xp = x + base;
    float*       op = x1 + base;
    for (int i = 0; i < CLEN; ++i) {
        float hv = hp[(size_t)i * DIMC];
        float nr = pr * sr - pi * si + hv;
        float ni = pr * si + pi * sr;
        sr = nr; si = ni;
        op[(size_t)i * DIMC] = sr * sc + xp[(size_t)i * DIMC];
    }
}

// ---------------------------------------------------------------------------
// Tile movers.
// A: [M][K] bf16 row-major,  Bt: [N][K] bf16 row-major (w1/w2 native layout).
// LDS tiles are 128 rows x 32 k with row stride LDSK=40 (conflict-free b128).
// ---------------------------------------------------------------------------
__device__ __forceinline__ void load_tile_A(const __bf16* __restrict__ src, int ldK,
                                            int rowBase, int k0, __bf16* __restrict__ dst) {
#pragma unroll
    for (int i = 0; i < 2; ++i) {
        int idx = threadIdx.x + 256 * i;        // 512 chunks of 8 bf16
        int row = idx >> 2;
        int kp  = (idx & 3) << 3;
        const __bf16* gp = src + (size_t)(rowBase + row) * ldK + (k0 + kp);
        __bf16* lp = dst + row * LDSK + kp;
#if HAVE_ASYNC_LDS
        // global_load_async_to_lds_b128: no VGPR staging, tracked by ASYNCcnt.
        // Param types per compiler diagnostic: AS1/AS3 pointers to int4.
        __builtin_amdgcn_global_load_async_to_lds_b128(
            (__attribute__((address_space(1))) v4i_vs*)(uintptr_t)gp,
            (__attribute__((address_space(3))) v4i_vs*)(uintptr_t)lp, 0, 0);
#else
        *(uint4*)lp = *(const uint4*)gp;        // global_load_b128 + ds_store_b128
#endif
    }
}

__device__ __forceinline__ void wait_async_tiles() {
#if HAVE_ASYNC_LDS
#  if defined(__has_builtin) && __has_builtin(__builtin_amdgcn_s_wait_asynccnt)
    __builtin_amdgcn_s_wait_asynccnt(0);
#  else
    asm volatile("s_wait_asynccnt 0" ::: "memory");
#  endif
#endif
}

// Same-shape fallback loader for the B tile (all 256 threads participate).
__device__ __forceinline__ void load_tile_B_manual(const __bf16* __restrict__ src, int ldK,
                                                   int rowBase, int k0, __bf16* __restrict__ dst) {
    load_tile_A(src, ldK, rowBase, k0, dst);
}

#if HAVE_TDM
// Tensor Data Mover: one 128x32 bf16 tile, LDS-padded to a 40-element stride
// (pad_interval = 16 DWORDs -> one 64B row, pad_amount = 4 DWORDs -> 16B pad).
// Issued by wave 0 only (TDM ignores EXEC); fenced with s_wait_tensorcnt.
__device__ __forceinline__ void tdm_load_tile_B(const __bf16* __restrict__ src, int ldK, int nTot,
                                                int rowBase, int k0, __bf16* __restrict__ dst) {
    const unsigned long long ga =
        (unsigned long long)(uintptr_t)(src + (size_t)rowBase * ldK + k0);
    const unsigned int lds = (unsigned int)(uintptr_t)dst;  // low 32 bits = LDS byte offset
    u32x4 g0;
    g0[0] = 1u;                                             // count=1, user descriptor
    g0[1] = lds;                                            // lds_addr
    g0[2] = (unsigned int)(ga & 0xFFFFFFFFu);               // global_addr[31:0]
    g0[3] = (unsigned int)((ga >> 32) & 0x01FFFFFFu)        // global_addr[56:32]
          | (2u << 30);                                     // type = 2 (image)
    i32x8 g1;
    g1[0] = (1 << 16)                                       // data_size = 2 bytes
          | (1 << 20)                                       // pad_enable
          | (3 << 22)                                       // pad_interval: 16 DWORDs
          | (3 << 25);                                      // pad_amount:   4 DWORDs
    g1[1] = (ldK & 0xFFFF) << 16;                           // tensor_dim0[15:0] = ldK
    g1[2] = (nTot & 0xFFFF) << 16;                          // tensor_dim1[15:0] = N
    g1[3] = 32 << 16;                                       // tile_dim0 = 32 elements
    g1[4] = 128;                                            // tile_dim1 = 128 rows
    g1[5] = ldK;                                            // tensor_dim0_stride = ldK
    g1[6] = 0;
    g1[7] = 0;
    i32x4 g2 = {0, 0, 0, 0};
    i32x4 g3 = {0, 0, 0, 0};
#  if __clang_major__ >= 23
    i32x8 g4 = {0, 0, 0, 0, 0, 0, 0, 0};
    __builtin_amdgcn_tensor_load_to_lds(g0, g1, g2, g3, g4, 0);
#  else
    __builtin_amdgcn_tensor_load_to_lds(g0, g1, g2, g3, 0);
#  endif
}
__device__ __forceinline__ void wait_tensor_tiles() {
    __builtin_amdgcn_s_wait_tensorcnt(0);
}
#endif

__device__ __forceinline__ bf16x16 frag16(const __bf16* __restrict__ p0,
                                          const __bf16* __restrict__ p1) {
    bf16x8 lo = *(const bf16x8*)p0;             // ds_load_b128
    bf16x8 hi = *(const bf16x8*)p1;             // ds_load_b128
    return __builtin_shufflevector(lo, hi, 0, 1, 2, 3, 4, 5, 6, 7,
                                           8, 9, 10, 11, 12, 13, 14, 15);
}

// ---------------------------------------------------------------------------
// WMMA GEMM mainloop body (shared by both GEMMs via macro to keep both
// epilogues explicit).  Block tile 128x128, 8 waves (2 M x 4 N), each wave
// 4x2 sub-tiles -> 8 v_wmma_f32_16x16x32_bf16 per 32-wide K step.
// ---------------------------------------------------------------------------
#if HAVE_TDM
#  define TDM_LOAD(BP, LDKV, NTOTV, RB, K0, DST)                                  \
      do { if (tdmWave) tdm_load_tile_B(BP, LDKV, NTOTV, RB, K0, DST); } while (0)
#  define TDM_WAIT()                                                              \
      do { if (tdmWave) wait_tensor_tiles(); } while (0)
#else
#  define TDM_LOAD(BP, LDKV, NTOTV, RB, K0, DST) do { } while (0)
#  define TDM_WAIT()                             do { } while (0)
#endif

#define GEMM_MAINLOOP(AP, BP, LDKV, NTOTV, KTV)                                   \
    load_tile_A(AP, LDKV, mBase, 0, sA[0]);                                       \
    if (useTdm) {                                                                 \
        TDM_LOAD(BP, LDKV, NTOTV, nBase, 0, sB[0]);                               \
    } else {                                                                      \
        load_tile_B_manual(BP, LDKV, nBase, 0, sB[0]);                            \
    }                                                                             \
    wait_async_tiles();                                                           \
    TDM_WAIT();                                                                   \
    __syncthreads();                                                              \
    for (int kt = 0; kt < (KTV); ++kt) {                                          \
        const int cur = kt & 1;                                                   \
        if (kt + 1 < (KTV)) {                                                     \
            load_tile_A(AP, LDKV, mBase, (kt + 1) * 32, sA[cur ^ 1]);             \
            if (useTdm) {                                                         \
                TDM_LOAD(BP, LDKV, NTOTV, nBase, (kt + 1) * 32, sB[cur ^ 1]);     \
            } else {                                                              \
                load_tile_B_manual(BP, LDKV, nBase, (kt + 1) * 32, sB[cur ^ 1]);  \
            }                                                                     \
        }                                                                         \
        if (kt + 2 < (KTV)) {                                                     \
            __builtin_prefetch(AP + (size_t)(mBase + (threadIdx.x >> 1)) * (LDKV) \
                                  + (kt + 2) * 32, 0, 0);                         \
            __builtin_prefetch(BP + (size_t)(nBase + (threadIdx.x >> 1)) * (LDKV) \
                                  + (kt + 2) * 32, 0, 0);                         \
        }                                                                         \
        const __bf16* __restrict__ As = sA[cur];                                  \
        const __bf16* __restrict__ Bs = sB[cur];                                  \
        bf16x16 af[4];                                                            \
        _Pragma("unroll")                                                         \
        for (int i = 0; i < 4; ++i) {                                             \
            const int row = wm * 64 + i * 16 + lm;                                \
            af[i] = frag16(As + row * LDSK + kbA, As + row * LDSK + kbA + 16);    \
        }                                                                         \
        bf16x16 bq[2];                                                            \
        _Pragma("unroll")                                                         \
        for (int j = 0; j < 2; ++j) {                                             \
            const int col = wn * 32 + j * 16 + lm;                                \
            bq[j] = frag16(Bs + col * LDSK + kbB, Bs + col * LDSK + kbB + 16);    \
        }                                                                         \
        _Pragma("unroll")                                                         \
        for (int i = 0; i < 4; ++i)                                               \
            _Pragma("unroll")                                                     \
            for (int j = 0; j < 2; ++j)                                           \
                acc[i][j] = __builtin_amdgcn_wmma_f32_16x16x32_bf16(              \
                    false, af[i], false, bq[j], (short)0, acc[i][j], false,       \
                    false);                                                       \
        wait_async_tiles();                                                       \
        TDM_WAIT();                                                               \
        __syncthreads();                                                          \
    }

// GEMM1: out[8192][4096] bf16 = silu(A[8192][1024] * w1^T + b1)
__global__ __launch_bounds__(256)
void gemm1_wmma_kernel(const __bf16* __restrict__ A, const __bf16* __restrict__ Bt,
                       const float* __restrict__ bias, __bf16* __restrict__ out) {
    __shared__ __bf16 sA[2][128 * LDSK];
    __shared__ __bf16 sB[2][128 * LDSK];
    const int mBase = blockIdx.y * 128;
    const int nBase = blockIdx.x * 128;
    const int lane = threadIdx.x & 31;
    const int w = threadIdx.x >> 5;
    const int wm = w & 1, wn = w >> 1;
    const int lm = lane & 15;
    const int kbA = (lane < 16) ? 0 : 8;
    const int kbB = (lane < 16) ? 0 : 8;
    const bool tdmWave = (threadIdx.x >> 5) == 0;   // whole wave 0, uniform
    const bool useTdm  = HAVE_TDM != 0;
    (void)tdmWave;
    f32x8 acc[4][2] = {};

    GEMM_MAINLOOP(A, Bt, DIMC, FFC, DIMC / 32)

    // epilogue: + bias, SiLU, bf16 store
#pragma unroll
    for (int i = 0; i < 4; ++i)
#pragma unroll
        for (int j = 0; j < 2; ++j) {
            const int col = nBase + wn * 32 + j * 16 + lm;
            const float bv = bias[col];
#pragma unroll
            for (int r = 0; r < 8; ++r) {
                const int row = mBase + wm * 64 + i * 16 + ((lane < 16) ? 0 : 8) + r;
                float v = acc[i][j][r] + bv;
                v = v / (1.0f + __expf(-v));
                out[(size_t)row * FFC + col] = (__bf16)v;
            }
        }
}

// GEMM2: out[8192][1024] f32 = A[8192][4096] * w2^T + b2 + x1
__global__ __launch_bounds__(256)
void gemm2_wmma_kernel(const __bf16* __restrict__ A, const __bf16* __restrict__ Bt,
                       const float* __restrict__ bias, const float* __restrict__ res,
                       float* __restrict__ out) {
    __shared__ __bf16 sA[2][128 * LDSK];
    __shared__ __bf16 sB[2][128 * LDSK];
    const int mBase = blockIdx.y * 128;
    const int nBase = blockIdx.x * 128;
    const int lane = threadIdx.x & 31;
    const int w = threadIdx.x >> 5;
    const int wm = w & 1, wn = w >> 1;
    const int lm = lane & 15;
    const int kbA = (lane < 16) ? 0 : 8;
    const int kbB = (lane < 16) ? 0 : 8;
    const bool tdmWave = (threadIdx.x >> 5) == 0;
    const bool useTdm  = HAVE_TDM != 0;
    (void)tdmWave;
    f32x8 acc[4][2] = {};

    GEMM_MAINLOOP(A, Bt, FFC, DIMC, FFC / 32)

    // epilogue: + bias + residual, f32 store
#pragma unroll
    for (int i = 0; i < 4; ++i)
#pragma unroll
        for (int j = 0; j < 2; ++j) {
            const int col = nBase + wn * 32 + j * 16 + lm;
            const float bv = bias[col];
#pragma unroll
            for (int r = 0; r < 8; ++r) {
                const int row = mBase + wm * 64 + i * 16 + ((lane < 16) ? 0 : 8) + r;
                out[(size_t)row * DIMC + col] =
                    acc[i][j][r] + bv + res[(size_t)row * DIMC + col];
            }
        }
}

// ---------------------------------------------------------------------------
// launch
// ---------------------------------------------------------------------------
extern "C" void kernel_launch(void* const* d_in, const int* in_sizes, int n_in,
                              void* d_out, int out_size, void* d_ws, size_t ws_size,
                              hipStream_t stream) {
    const float* x       = (const float*)d_in[0];
    const float* ln1_w   = (const float*)d_in[1];
    const float* ln1_b   = (const float*)d_in[2];
    const float* ln2_w   = (const float*)d_in[3];
    const float* ln2_b   = (const float*)d_in[4];
    const float* sc_lin  = (const float*)d_in[5];
    const float* ph_re   = (const float*)d_in[6];
    const float* ph_im   = (const float*)d_in[7];
    const float* last_re = (const float*)d_in[8];
    const float* last_im = (const float*)d_in[9];
    const float* w1      = (const float*)d_in[10];
    const float* b1      = (const float*)d_in[11];
    const float* w2      = (const float*)d_in[12];
    const float* b2      = (const float*)d_in[13];
    float* out = (float*)d_out;

    // workspace layout (~161 MiB)
    char* ws = (char*)d_ws;
    size_t off = 0;
    __bf16* w1bf  = (__bf16*)(ws + off); off += (size_t)FFC * DIMC * 2;     // 8 MiB
    __bf16* w2bf  = (__bf16*)(ws + off); off += (size_t)DIMC * FFC * 2;     // 8 MiB
    float*  h     = (float*) (ws + off); off += (size_t)ROWSC * DIMC * 4;   // 32 MiB
    float*  x1    = (float*) (ws + off); off += (size_t)ROWSC * DIMC * 4;   // 32 MiB
    __bf16* h2    = (__bf16*)(ws + off); off += (size_t)ROWSC * DIMC * 2;   // 16 MiB
    __bf16* a1    = (__bf16*)(ws + off); off += (size_t)ROWSC * FFC * 2;    // 64 MiB
    float*  cS    = (float*) (ws + off); off += (size_t)BATC * NCH * DIMC * 2 * 4;
    float*  cI    = (float*) (ws + off); off += (size_t)BATC * NCH * DIMC * 2 * 4;
    (void)off; (void)ws_size; (void)in_sizes; (void)n_in; (void)out_size;

    // 0) weights -> bf16 (layout [N][K] already matches the B-tile layout)
    cvt_bf16_kernel<<<2048, 256, 0, stream>>>(w1, w1bf, FFC * DIMC);
    cvt_bf16_kernel<<<2048, 256, 0, stream>>>(w2, w2bf, DIMC * FFC);

    // 1) h = silu(LN1(x))
    ln1_silu_kernel<<<ROWSC, 256, 0, stream>>>(x, ln1_w, ln1_b, h);

    // 2) SConv chunked scan -> x1 = Re(scan)*sc_lin + x
    scan_pass1_kernel<<<(BATC * NCH * DIMC) / 256, 256, 0, stream>>>(h, ph_re, ph_im, cS);
    scan_pass2_kernel<<<(BATC * DIMC) / 256, 256, 0, stream>>>(ph_re, ph_im, last_re, last_im, cS, cI);
    scan_pass3_kernel<<<(BATC * NCH * DIMC) / 256, 256, 0, stream>>>(h, x, ph_re, ph_im, sc_lin, cI, x1);

    // 3) h2 = bf16(LN2(x1))
    ln2_bf16_kernel<<<ROWSC, 256, 0, stream>>>(x1, ln2_w, ln2_b, h2);

    // 4) a1 = bf16(silu(h2 @ w1^T + b1))   [8192 x 4096]
    gemm1_wmma_kernel<<<dim3(FFC / 128, ROWSC / 128), 256, 0, stream>>>(h2, w1bf, b1, a1);

    // 5) out = a1 @ w2^T + b2 + x1         [8192 x 1024]
    gemm2_wmma_kernel<<<dim3(DIMC / 128, ROWSC / 128), 256, 0, stream>>>(a1, w2bf, b2, x1, out);
}
